// GCN_3_38963943309622
// MI455X (gfx1250) — matrix operand order repaired and checked by reading, not compiled, not run on hardware
//
#include <hip/hip_runtime.h>

// ---------------------------------------------------------------------------
// GCN-3 for MI455X (gfx1250, wave32).  GEMMs run on v_wmma_f32_16x16x32_bf16;
// edge aggregation is fp32 global_atomic_add_f32 (node features fit in 192MB L2).
// ---------------------------------------------------------------------------

#define NN   50000      // nodes
#define NE   800000     // edges
#define KD   128        // in/hidden dim (K of every GEMM)
#define FH   128        // hidden width
#define FC   40         // output classes
#define FCP  48         // padded output width (3 x 16-col WMMA tiles)

typedef __attribute__((ext_vector_type(16))) __bf16 v16bf;
typedef __attribute__((ext_vector_type(8)))  float  v8f;

union Frag16 { v16bf v; uint4 q[2]; };

// round-to-nearest-even f32 -> bf16 (stored as u16; WMMA sees raw bits)
__device__ __forceinline__ unsigned short f2bf(float f) {
  unsigned int u = __float_as_uint(f);
  u += 0x7FFFu + ((u >> 16) & 1u);
  return (unsigned short)(u >> 16);
}

// ---------------------------------------------------------------- preprocess
__global__ void k_zero_f32(float* __restrict__ p, int n) {
  int i = blockIdx.x * blockDim.x + threadIdx.x;
  if (i < n) p[i] = 0.0f;
}

__global__ void k_deg_count(const int* __restrict__ ei, float* __restrict__ deg) {
  int e = blockIdx.x * blockDim.x + threadIdx.x;
  if (e < NE) atomicAdd(&deg[ei[NE + e]], 1.0f);   // dst row of edge_index
}

__global__ void k_make_dinv(float* __restrict__ d) {   // in: deg, out: dinv
  int i = blockIdx.x * blockDim.x + threadIdx.x;
  if (i < NN) d[i] = rsqrtf(d[i] + 1.0f);              // +1 self loop, >= 1
}

__global__ void k_f32_to_bf16(const float* __restrict__ s,
                              unsigned short* __restrict__ d, int n) {
  int i = blockIdx.x * blockDim.x + threadIdx.x;
  if (i < n) d[i] = f2bf(s[i]);
}

// W[k][f] (row-major KxF) -> Wt[f][k] bf16, zero-padded to Fpad rows
__global__ void k_w_transpose_bf16(const float* __restrict__ W,
                                   unsigned short* __restrict__ Wt,
                                   int F, int Fpad) {
  int i = blockIdx.x * blockDim.x + threadIdx.x;
  if (i >= Fpad * KD) return;
  int f = i / KD, k = i - f * KD;
  float v = (f < F) ? W[k * F + f] : 0.0f;
  Wt[i] = f2bf(v);
}

// ------------------------------------------------------------------ GEMM
// C[N x F] = A[N x 128](bf16) * B[128 x F]   (B given transposed: Bt[F][128])
// one block = 16-row tile; wave w = 16-col tile; 4 WMMA steps over K=128.
__global__ __launch_bounds__(256)
void k_gemm_wmma_bf16(const unsigned short* __restrict__ A,
                      const unsigned short* __restrict__ Bt,
                      float* __restrict__ C, int F) {
  const int wave = threadIdx.x >> 5;
  const int lane = threadIdx.x & 31;
  const int half = lane >> 4;       // lane-half select per ISA fragment layout
  const int l16  = lane & 15;
  const int r0   = blockIdx.x << 4; // N = 3125 * 16 exactly, no tail
  const int c0   = wave << 4;

  const unsigned short* aRow = A  + (size_t)(r0 + l16) * KD;  // A lane row = M
  const unsigned short* bCol = Bt + (size_t)(c0 + l16) * KD;  // B lane row = N

  v8f acc = {};
#pragma unroll
  for (int s = 0; s < 4; ++s) {
    const int kb = s << 5;
    Frag16 a, b;
    // A 16x32 bf16: elem e<8 -> K = kb + half*8 + e ; e>=8 -> K = kb+16+half*8+(e-8)
    a.q[0] = *(const uint4*)(aRow + kb + half * 8);
    a.q[1] = *(const uint4*)(aRow + kb + 16 + half * 8);
    // B 32x16 bf16: elem e<8 -> K = kb + half*8 + e ; e>=8 -> K = kb+16+half*8+(e-8)
    b.q[0] = *(const uint4*)(bCol + kb + half * 8);
    b.q[1] = *(const uint4*)(bCol + kb + 16 + half * 8);
    acc = __builtin_amdgcn_wmma_f32_16x16x32_bf16(
        /*neg_a=*/false, a.v, /*neg_b=*/false, b.v,
        /*c_mod=*/(short)0, acc, /*reuse_a=*/false, /*reuse_b=*/false);
  }
  // C/D layout: vgpr v -> M = v + half*8 ; lane&15 -> N
#pragma unroll
  for (int v = 0; v < 8; ++v)
    C[(size_t)(r0 + v + half * 8) * F + c0 + l16] = acc[v];
}

// ------------------------------------------------------------- aggregation
// agg[i][f] = t[i][f] * dinv[i]^2 + bias[f]         (self-loop + bias)
__global__ void k_init_agg128(const float* __restrict__ t,
                              const float* __restrict__ dinv,
                              const float* __restrict__ bias,
                              float* __restrict__ agg) {
  int i = blockIdx.x * blockDim.x + threadIdx.x;
  if (i >= NN * FH) return;
  int node = i >> 7, f = i & 127;
  float di = dinv[node];
  agg[i] = t[i] * di * di + bias[f];
}

// scatter: agg[dst] += dinv[src]*dinv[dst] * t[src]   (float4 per thread)
__global__ void k_scatter128(const int* __restrict__ ei,
                             const float* __restrict__ dinv,
                             const float* __restrict__ t,
                             float* __restrict__ agg) {
  int tid = blockIdx.x * blockDim.x + threadIdx.x;
  if (tid >= NE * 32) return;
  int e = tid >> 5, q = tid & 31;
  int src = ei[e], dst = ei[NE + e];
  float nrm = dinv[src] * dinv[dst];
  float4 v = ((const float4*)(t + (size_t)src * FH))[q];
  float* o = agg + (size_t)dst * FH + q * 4;
  atomicAdd(o + 0, nrm * v.x);
  atomicAdd(o + 1, nrm * v.y);
  atomicAdd(o + 2, nrm * v.z);
  atomicAdd(o + 3, nrm * v.w);
}

// ReLU + bf16 convert -> next layer input
__global__ void k_relu_cvt128(const float* __restrict__ agg,
                              unsigned short* __restrict__ hb) {
  int i = blockIdx.x * blockDim.x + threadIdx.x;
  if (i < NN * FH) hb[i] = f2bf(fmaxf(agg[i], 0.0f));
}

// final layer: out stride 40, GEMM result t stride 48 (padded)
__global__ void k_init_out40(const float* __restrict__ t,
                             const float* __restrict__ dinv,
                             const float* __restrict__ bias,
                             float* __restrict__ out) {
  int i = blockIdx.x * blockDim.x + threadIdx.x;
  if (i >= NN * FC) return;
  int node = i / FC, c = i - node * FC;
  float di = dinv[node];
  out[i] = t[(size_t)node * FCP + c] * di * di + bias[c];
}

__global__ void k_scatter40(const int* __restrict__ ei,
                            const float* __restrict__ dinv,
                            const float* __restrict__ t,
                            float* __restrict__ out) {
  int tid = blockIdx.x * blockDim.x + threadIdx.x;
  if (tid >= NE * FC) return;
  int e = tid / FC, c = tid - e * FC;
  int src = ei[e], dst = ei[NE + e];
  float nrm = dinv[src] * dinv[dst];
  atomicAdd(out + (size_t)dst * FC + c, nrm * t[(size_t)src * FCP + c]);
}

// ------------------------------------------------------------------ driver
extern "C" void kernel_launch(void* const* d_in, const int* in_sizes, int n_in,
                              void* d_out, int out_size, void* d_ws, size_t ws_size,
                              hipStream_t stream) {
  const float* x  = (const float*)d_in[0];
  const int*   ei = (const int*)  d_in[1];
  const float* W1 = (const float*)d_in[2];
  const float* b1 = (const float*)d_in[3];
  const float* W2 = (const float*)d_in[4];
  const float* b2 = (const float*)d_in[5];
  const float* W3 = (const float*)d_in[6];
  const float* b3 = (const float*)d_in[7];
  float* out = (float*)d_out;

  // workspace carve-out (~64.3 MB total)
  char* ws = (char*)d_ws;
  size_t off = 0;
  auto carve = [&](size_t bytes) -> void* {
    void* p = ws + off;
    off += (bytes + 255) & ~(size_t)255;
    return p;
  };
  float*          dinv = (float*)         carve((size_t)NN * 4);
  unsigned short* hb   = (unsigned short*)carve((size_t)NN * KD * 2);
  float*          tbuf = (float*)         carve((size_t)NN * KD * 4);
  float*          agg  = (float*)         carve((size_t)NN * KD * 4);
  unsigned short* Wt1  = (unsigned short*)carve((size_t)FH * KD * 2);
  unsigned short* Wt2  = (unsigned short*)carve((size_t)FH * KD * 2);
  unsigned short* Wt3  = (unsigned short*)carve((size_t)FCP * KD * 2);

  const int T = 256;
  auto g = [](long long n, int t) { return (unsigned)((n + t - 1) / t); };

  // --- graph normalization ---
  k_zero_f32 <<<g(NN, T), T, 0, stream>>>(dinv, NN);
  k_deg_count<<<g(NE, T), T, 0, stream>>>(ei, dinv);
  k_make_dinv<<<g(NN, T), T, 0, stream>>>(dinv);

  // --- bf16 precompute ---
  k_f32_to_bf16<<<g((long long)NN * KD, T), T, 0, stream>>>(x, hb, NN * KD);
  k_w_transpose_bf16<<<g(FH  * KD, T), T, 0, stream>>>(W1, Wt1, FH, FH);
  k_w_transpose_bf16<<<g(FH  * KD, T), T, 0, stream>>>(W2, Wt2, FH, FH);
  k_w_transpose_bf16<<<g(FCP * KD, T), T, 0, stream>>>(W3, Wt3, FC, FCP);

  const unsigned rowTiles = NN / 16;   // 3125, exact

  // --- layer 1 ---
  k_gemm_wmma_bf16<<<rowTiles, 256, 0, stream>>>(hb, Wt1, tbuf, FH);
  k_init_agg128  <<<g((long long)NN * FH, T), T, 0, stream>>>(tbuf, dinv, b1, agg);
  k_scatter128   <<<g((long long)NE * 32, T), T, 0, stream>>>(ei, dinv, tbuf, agg);
  k_relu_cvt128  <<<g((long long)NN * FH, T), T, 0, stream>>>(agg, hb);

  // --- layer 2 ---
  k_gemm_wmma_bf16<<<rowTiles, 256, 0, stream>>>(hb, Wt2, tbuf, FH);
  k_init_agg128  <<<g((long long)NN * FH, T), T, 0, stream>>>(tbuf, dinv, b2, agg);
  k_scatter128   <<<g((long long)NE * 32, T), T, 0, stream>>>(ei, dinv, tbuf, agg);
  k_relu_cvt128  <<<g((long long)NN * FH, T), T, 0, stream>>>(agg, hb);

  // --- layer 3 (no ReLU, scatter straight into d_out) ---
  k_gemm_wmma_bf16<<<rowTiles, 96, 0, stream>>>(hb, Wt3, tbuf, FCP);
  k_init_out40   <<<g((long long)NN * FC, T), T, 0, stream>>>(tbuf, dinv, b3, out);
  k_scatter40    <<<g((long long)NE * FC, T), T, 0, stream>>>(ei, dinv, tbuf, out);
}
